// MyOutRGAT_687194767722
// MI455X (gfx1250) — compile-verified, matrix operand order checked by reading, not run on hardware
//
#include <hip/hip_runtime.h>
#include <hip/hip_bf16.h>
#include <stdint.h>

#define DEVINL __device__ __forceinline__

typedef __attribute__((ext_vector_type(16))) __bf16 v16bf;
typedef __attribute__((ext_vector_type(8)))  float  v8f;

union Frag {
    uint4  q[2];
    unsigned u[8];
    v16bf  v;
};
union Acc {
    v8f   v;
    float f[8];
};

DEVINL unsigned f2bf(float x) {
    unsigned b = __float_as_uint(x);
    return (b + 0x7FFFu + ((b >> 16) & 1u)) >> 16;   // RNE truncate to bf16
}
DEVINL float ordDec(unsigned o) {
    return (o & 0x80000000u) ? __uint_as_float(o ^ 0x80000000u)
                             : __uint_as_float(~o);
}
DEVINL unsigned ordEnc(float x) {
    unsigned b = __float_as_uint(x);
    return (b & 0x80000000u) ? ~b : (b | 0x80000000u);
}
DEVINL void atomAddF(float* p, float v) { unsafeAtomicAdd(p, v); }

// bf16 A/B fragment K-pair base per VGPR index v and lane-half (ISA 7.12.2)
DEVINL int kBase(int v, int hi) {
    return (v < 4) ? ((hi ? 8 : 0) + 2 * v) : ((hi ? 24 : 16) + 2 * (v - 4));
}

// ---------------------------------------------------------------- utilities
__global__ void fill_u32_k(unsigned* p, unsigned val, long long n) {
    long long i = (long long)blockIdx.x * blockDim.x + threadIdx.x;
    long long s = (long long)gridDim.x * blockDim.x;
    for (; i < n; i += s) p[i] = val;
}

__global__ void cvt_bf16_k(const float* __restrict__ src,
                           unsigned short* __restrict__ dst, long long n) {
    long long i = (long long)blockIdx.x * blockDim.x + threadIdx.x;
    long long s = (long long)gridDim.x * blockDim.x;
    for (; i < n; i += s) dst[i] = (unsigned short)f2bf(src[i]);
}

// Repack W[r][k][col] (fp32) into per-lane WMMA B-fragment dwords:
// index = (((r*NB + colBlk)*KS + kStep)*32 + lane)*8 + v
__global__ void repack_w_k(const float* __restrict__ W, unsigned* __restrict__ WB,
                           int K, int Cout) {
    int NB = Cout >> 4, KS = K >> 5;
    int total = 2 * NB * KS * 256;
    int t = blockIdx.x * blockDim.x + threadIdx.x;
    if (t >= total) return;
    int v = t & 7;
    int lane = (t >> 3) & 31;
    int rest = t >> 8;
    int ks = rest % KS; rest /= KS;
    int cb = rest % NB;
    int r  = rest / NB;
    int col = cb * 16 + (lane & 15);
    int k = ks * 32 + kBase(v, lane >> 4);
    const float* Wr = W + (size_t)r * K * Cout;
    unsigned lo = f2bf(Wr[(size_t)k * Cout + col]);
    unsigned hi = f2bf(Wr[(size_t)(k + 1) * Cout + col]);
    WB[t] = lo | (hi << 16);
}

// ------------------------------------------------------------- WMMA GEMM
// XW[r] = Xb @ W[r];  Xb: [N,K] bf16 row-major, WB: repacked fragments,
// XW: [R,N,Cout] fp32.  Wave -> 32x32 tile; block = 8 waves = 256 rows.
__global__ void gemm_bf16_k(const unsigned short* __restrict__ Xb,
                            const unsigned* __restrict__ WB,
                            float* __restrict__ XW,
                            int N, int K, int Cout) {
    const int lane = threadIdx.x & 31;
    const int wv   = threadIdx.x >> 5;
    const int r    = blockIdx.z;
    const int m0   = blockIdx.x * 256 + wv * 32;
    const int n0   = blockIdx.y * 32;
    const int hi   = lane >> 4;
    const int lr   = lane & 15;
    const int KS   = K >> 5;
    const int NB   = Cout >> 4;

    int rowA0 = m0 + lr;       if (rowA0 >= N) rowA0 = N - 1;
    int rowA1 = m0 + 16 + lr;  if (rowA1 >= N) rowA1 = N - 1;
    // lanes 0-15 need K chunks {0..7},{16..23}; lanes 16-31 {8..15},{24..31}
    const unsigned short* A0p = Xb + (size_t)rowA0 * K + (hi ? 8 : 0);
    const unsigned short* A1p = Xb + (size_t)rowA1 * K + (hi ? 8 : 0);
    const unsigned* B0p = WB + (((size_t)r * NB + (n0 >> 4)) * KS) * 256 + lane * 8;
    const unsigned* B1p = WB + (((size_t)r * NB + (n0 >> 4) + 1) * KS) * 256 + lane * 8;

    Acc c00, c01, c10, c11;
#pragma unroll
    for (int i = 0; i < 8; ++i) { c00.f[i] = 0.f; c01.f[i] = 0.f; c10.f[i] = 0.f; c11.f[i] = 0.f; }

    for (int ks = 0; ks < KS; ++ks) {
        Frag a0, a1, b0, b1;
        const unsigned short* pa0 = A0p + ks * 32;
        const unsigned short* pa1 = A1p + ks * 32;
        a0.q[0] = *(const uint4*)(pa0);
        a0.q[1] = *(const uint4*)(pa0 + 16);
        a1.q[0] = *(const uint4*)(pa1);
        a1.q[1] = *(const uint4*)(pa1 + 16);
        b0.q[0] = *(const uint4*)(B0p + (size_t)ks * 256);
        b0.q[1] = *(const uint4*)(B0p + (size_t)ks * 256 + 4);
        b1.q[0] = *(const uint4*)(B1p + (size_t)ks * 256);
        b1.q[1] = *(const uint4*)(B1p + (size_t)ks * 256 + 4);
        c00.v = __builtin_amdgcn_wmma_f32_16x16x32_bf16(false, a0.v, false, b0.v, (short)0, c00.v, false, false);
        c01.v = __builtin_amdgcn_wmma_f32_16x16x32_bf16(false, a0.v, false, b1.v, (short)0, c01.v, false, false);
        c10.v = __builtin_amdgcn_wmma_f32_16x16x32_bf16(false, a1.v, false, b0.v, (short)0, c10.v, false, false);
        c11.v = __builtin_amdgcn_wmma_f32_16x16x32_bf16(false, a1.v, false, b1.v, (short)0, c11.v, false, false);
    }

    float* out = XW + (size_t)r * N * Cout;
#pragma unroll
    for (int v = 0; v < 8; ++v) {
        int mr  = hi ? (8 + v) : v;           // C/D layout: lanes<16 M=v, else M=8+v
        int gr0 = m0 + mr;
        int gr1 = m0 + 16 + mr;
        if (gr0 < N) {
            out[(size_t)gr0 * Cout + n0 + lr]      = c00.f[v];
            out[(size_t)gr0 * Cout + n0 + 16 + lr] = c01.f[v];
        }
        if (gr1 < N) {
            out[(size_t)gr1 * Cout + n0 + lr]      = c10.f[v];
            out[(size_t)gr1 * Cout + n0 + 16 + lr] = c11.f[v];
        }
    }
}

// ---------------------------------------------------- per-node attn scalars
// sI[r][n][h] = xw[r][n]·q[:,h]; sJ[r][n][h] = xw[r][n]·k[:,h]
__global__ void scores_k(const float* __restrict__ XW,
                         const float* __restrict__ q, const float* __restrict__ kk,
                         float* __restrict__ sI, float* __restrict__ sJ,
                         int N, int C, int H) {
    int lane = threadIdx.x & 31;
    int gw = blockIdx.x * 8 + (threadIdx.x >> 5);
    if (gw >= 2 * N) return;
    int r = gw / N, n = gw % N;
    const float* row = XW + ((size_t)r * N + n) * C;
    float aI[2] = {0.f, 0.f}, aJ[2] = {0.f, 0.f};
    for (int c = lane; c < C; c += 32) {
        float v = row[c];
        for (int h = 0; h < H; ++h) {
            aI[h] += v * q[c * H + h];
            aJ[h] += v * kk[c * H + h];
        }
    }
    for (int off = 16; off > 0; off >>= 1)
        for (int h = 0; h < H; ++h) {
            aI[h] += __shfl_xor(aI[h], off, 32);
            aJ[h] += __shfl_xor(aJ[h], off, 32);
        }
    if (lane == 0)
        for (int h = 0; h < H; ++h) {
            sI[((size_t)r * N + n) * H + h] = aI[h];
            sJ[((size_t)r * N + n) * H + h] = aJ[h];
        }
}

// ------------------------------------------------------------- edge passes
__global__ void edge_logit_max_k(const int* __restrict__ ei, const int* __restrict__ et,
                                 const float* __restrict__ sI, const float* __restrict__ sJ,
                                 float* __restrict__ alpha, unsigned* __restrict__ menc,
                                 int N, int E, int H) {
    long long t = (long long)blockIdx.x * blockDim.x + threadIdx.x;
    if (t >= (long long)E * H) return;
    int e = (int)(t / H), h = (int)(t % H);
    int src = ei[e], dst = ei[E + e], ty = et[e];
    float a = sI[((size_t)ty * N + dst) * H + h] + sJ[((size_t)ty * N + src) * H + h];
    a = (a > 0.f) ? a : 0.2f * a;                 // leaky_relu(., 0.2)
    alpha[t] = a;
    atomicMax(&menc[(size_t)dst * H + h], ordEnc(a));
}

__global__ void edge_expsum_k(const int* __restrict__ ei,
                              float* __restrict__ alpha,
                              const unsigned* __restrict__ menc,
                              float* __restrict__ ssum, int E, int H) {
    long long t = (long long)blockIdx.x * blockDim.x + threadIdx.x;
    if (t >= (long long)E * H) return;
    int e = (int)(t / H), h = (int)(t % H);
    int dst = ei[E + e];
    float m = ordDec(menc[(size_t)dst * H + h]);
    float ev = __expf(alpha[t] - m);
    alpha[t] = ev;
    atomAddF(&ssum[(size_t)dst * H + h], ev);
}

__global__ void edge_message_k(const int* __restrict__ ei, const int* __restrict__ et,
                               const float* __restrict__ alpha, const float* __restrict__ ssum,
                               const float* __restrict__ XW, float* __restrict__ hbuf,
                               int N, int E, int C, int H) {
    long long t = (long long)blockIdx.x * blockDim.x + threadIdx.x;
    if (t >= (long long)E * C) return;
    int e = (int)(t / C), c = (int)(t % C);
    int src = ei[e], dst = ei[E + e], ty = et[e];
    int h = c / (C / H);
    float an = alpha[(size_t)e * H + h] / (ssum[(size_t)dst * H + h] + 1e-16f);
    float v = an * XW[((size_t)ty * N + src) * C + c];
    atomAddF(&hbuf[(size_t)dst * C + c], v);
}

// ------------------------------------------- bias + leakyrelu(0.01) + stats
__global__ void finalize_stats_k(float* __restrict__ hbuf, const float* __restrict__ b,
                                 float* __restrict__ stats, int N, int C) {
    int c = threadIdx.x % C;
    int which = threadIdx.x / C;
    int lanesPer = 256 / C;
    int base = blockIdx.x * 64;
    float s = 0.f, sq = 0.f;
    for (int n = base + which; n < base + 64 && n < N; n += lanesPer) {
        float v = hbuf[(size_t)n * C + c] + b[c];
        v = (v > 0.f) ? v : 0.01f * v;
        hbuf[(size_t)n * C + c] = v;
        s += v; sq += v * v;
    }
    atomAddF(&stats[c], s);
    atomAddF(&stats[C + c], sq);
}

// BatchNorm apply: mode 0 -> write bf16 (next layer input); mode 1 -> fp32 in-place
__global__ void bn_apply_k(const float* __restrict__ hin, const float* __restrict__ stats,
                           const float* __restrict__ g, const float* __restrict__ be,
                           unsigned short* __restrict__ obf, float* __restrict__ of32,
                           int N, int C, int mode) {
    long long t = (long long)blockIdx.x * blockDim.x + threadIdx.x;
    if (t >= (long long)N * C) return;
    int c = (int)(t % C);
    float mu = stats[c] / (float)N;
    float var = stats[C + c] / (float)N - mu * mu;
    float y = g[c] * (hin[t] - mu) * rsqrtf(var + 1e-5f) + be[c];
    if (mode == 0) obf[t] = (unsigned short)f2bf(y);
    else           of32[t] = y;
}

// ------------------------------------- head: gather + [128x5] GEMV + sigmoid
__global__ void head_k(const float* __restrict__ h1, const int* __restrict__ idx,
                       const float* __restrict__ wm, const float* __restrict__ bm,
                       float* __restrict__ out, int Ksel) {
    __shared__ float sh[128];
    int row = blockIdx.x;
    int c = threadIdx.x;
    float v = h1[(size_t)idx[row] * 128 + c];
    out[(size_t)row * 128 + c] = v;
    sh[c] = v;
    __syncthreads();
    if (c < 5) {
        float acc = bm[c];
        for (int i = 0; i < 128; ++i) acc += sh[i] * wm[i * 5 + c];
        out[(size_t)Ksel * 128 + (size_t)row * 5 + c] = 1.f / (1.f + __expf(-acc));
    }
}

// ------------------------------------------------------------------- host
static void fillU32(unsigned* p, unsigned v, long long n, hipStream_t s) {
    int blocks = (int)(((n + 255) / 256) < 65535 ? ((n + 255) / 256) : 65535);
    fill_u32_k<<<blocks, 256, 0, s>>>(p, v, n);
}

static void run_layer(const unsigned short* xb_in, unsigned short* xb_out,
                      const float* W, const float* q, const float* kk,
                      const float* b, const float* g, const float* be,
                      const int* ei, const int* et,
                      unsigned* wb, float* xw, float* sI, float* sJ,
                      unsigned* menc, float* ssum, float* alpha,
                      float* hbuf, float* stats,
                      int N, int E, int Cin, int Cout, int H, int final_fp32,
                      hipStream_t stream) {
    int NB = Cout >> 4, KS = Cin >> 5;
    int repackTot = 2 * NB * KS * 256;
    repack_w_k<<<(repackTot + 255) / 256, 256, 0, stream>>>(W, wb, Cin, Cout);

    dim3 gg((N + 255) / 256, Cout / 32, 2);
    gemm_bf16_k<<<gg, 256, 0, stream>>>(xb_in, wb, xw, N, Cin, Cout);

    scores_k<<<(2 * N + 7) / 8, 256, 0, stream>>>(xw, q, kk, sI, sJ, N, Cout, H);

    fillU32(menc, 0x007FFFFFu /* ordEnc(-inf) */, (long long)N * H, stream);
    fillU32((unsigned*)ssum, 0u, (long long)N * H, stream);
    fillU32((unsigned*)hbuf, 0u, (long long)N * Cout, stream);
    fillU32((unsigned*)stats, 0u, 2 * Cout, stream);

    long long eh = (long long)E * H;
    edge_logit_max_k<<<(int)((eh + 255) / 256), 256, 0, stream>>>(ei, et, sI, sJ, alpha, menc, N, E, H);
    edge_expsum_k<<<(int)((eh + 255) / 256), 256, 0, stream>>>(ei, alpha, menc, ssum, E, H);

    long long ec = (long long)E * Cout;
    edge_message_k<<<(int)((ec + 255) / 256), 256, 0, stream>>>(ei, et, alpha, ssum, xw, hbuf, N, E, Cout, H);

    finalize_stats_k<<<(N + 63) / 64, 256, 0, stream>>>(hbuf, b, stats, N, Cout);

    long long nc = (long long)N * Cout;
    bn_apply_k<<<(int)((nc + 255) / 256), 256, 0, stream>>>(hbuf, stats, g, be, xb_out, hbuf, N, Cout, final_fp32);
}

extern "C" void kernel_launch(void* const* d_in, const int* in_sizes, int n_in,
                              void* d_out, int out_size, void* d_ws, size_t ws_size,
                              hipStream_t stream) {
    (void)in_sizes; (void)n_in; (void)out_size; (void)ws_size;
    const float* x   = (const float*)d_in[0];
    const int*   ei  = (const int*)d_in[1];
    const int*   et  = (const int*)d_in[2];
    const int*   idx = (const int*)d_in[3];
    const float* w0  = (const float*)d_in[4];
    const float* q0  = (const float*)d_in[5];
    const float* k0  = (const float*)d_in[6];
    const float* b0  = (const float*)d_in[7];
    const float* g0  = (const float*)d_in[8];
    const float* be0 = (const float*)d_in[9];
    const float* w1  = (const float*)d_in[10];
    const float* q1  = (const float*)d_in[11];
    const float* k1  = (const float*)d_in[12];
    const float* b1  = (const float*)d_in[13];
    const float* g1  = (const float*)d_in[14];
    const float* be1 = (const float*)d_in[15];
    const float* wm  = (const float*)d_in[16];
    const float* bm  = (const float*)d_in[17];
    float* out = (float*)d_out;

    const int N = 50000, E = 800000, KSEL = 4096;

    char* p = (char*)d_ws;
    auto take = [&](size_t bytes) -> char* {
        char* r = p; p += (bytes + 255) & ~(size_t)255; return r;
    };
    unsigned short* xb   = (unsigned short*)take((size_t)N * 256 * 2);
    unsigned*       wb   = (unsigned*)      take((size_t)32768 * 4);
    float*          xw   = (float*)         take((size_t)2 * N * 256 * 4);
    float*          sI   = (float*)         take((size_t)2 * N * 2 * 4);
    float*          sJ   = (float*)         take((size_t)2 * N * 2 * 4);
    unsigned*       menc = (unsigned*)      take((size_t)N * 2 * 4);
    float*          ssum = (float*)         take((size_t)N * 2 * 4);
    float*          alph = (float*)         take((size_t)E * 2 * 4);
    float*          hbuf = (float*)         take((size_t)N * 256 * 4);
    float*          stat = (float*)         take(2048);

    // x fp32 -> bf16 for layer-0 GEMM
    cvt_bf16_k<<<4096, 256, 0, stream>>>(x, xb, (long long)N * 128);

    // Layer 0: in=128, out=256 (2 heads x 128); BN output -> bf16 xb
    run_layer(xb, xb, w0, q0, k0, b0, g0, be0, ei, et,
              wb, xw, sI, sJ, menc, ssum, alph, hbuf, stat,
              N, E, 128, 256, 2, 0, stream);

    // Layer 1: in=256, out=128 (1 head); BN output -> fp32 in hbuf
    run_layer(xb, xb, w1, q1, k1, b1, g1, be1, ei, et,
              wb, xw, sI, sJ, menc, ssum, alph, hbuf, stat,
              N, E, 256, 128, 1, 1, stream);

    // Gather idx rows, write h, compute sigmoid head
    head_k<<<KSEL, 128, 0, stream>>>(hbuf, idx, wm, bm, out, KSEL);
}
